// Encoder_36051955482724
// MI455X (gfx1250) — compile-verified, matrix operand order checked by reading, not run on hardware
//
#include <hip/hip_runtime.h>
#include <math.h>

// ---------------------------------------------------------------------------
// Encoder: embed + BN + GRU(400) + GRU(200), B=128, T=256, EMB=200.
// WMMA f16 (v_wmma_f32_16x16x32_f16) for all matmuls, f32 accumulate.
// Recurrence = persistent kernels with per-step device barrier + double-
// buffered hidden state; recurrent weights staged into LDS via the Tensor
// Data Mover (tensor_load_to_lds + s_wait_tensorcnt) when available.
// ---------------------------------------------------------------------------

typedef _Float16 v8h   __attribute__((ext_vector_type(8)));
typedef _Float16 v16h  __attribute__((ext_vector_type(16)));
typedef float    v8f   __attribute__((ext_vector_type(8)));
typedef unsigned u32x4 __attribute__((ext_vector_type(4)));
typedef int      i32x4 __attribute__((ext_vector_type(4)));
typedef int      i32x8 __attribute__((ext_vector_type(8)));

static constexpr int kB    = 128;
static constexpr int kT    = 256;
static constexpr int kEMB  = 200;
static constexpr int kU1   = 400;
static constexpr int kU2   = 200;
static constexpr int kKp1  = 224;      // 200 padded to 7*32
static constexpr int kKp2  = 416;      // 400 padded to 13*32
static constexpr int kRows = kB * kT;  // 32768
static constexpr int kN1   = 1200;     // 3*U1
static constexpr int kN2p  = 608;      // 3*U2=600 padded to 38*16

#define DEVINL __device__ __forceinline__

#if defined(__has_builtin)
#if __has_builtin(__builtin_amdgcn_tensor_load_to_lds) && \
    __has_builtin(__builtin_amdgcn_s_wait_tensorcnt)
#define HAVE_TDM 1
#endif
#endif
#ifndef HAVE_TDM
#define HAVE_TDM 0
#endif

// Load one 16x32 f16 WMMA fragment (A layout; B uses same pattern on
// transposed weights). lane&15 = row, (lane>>4)*8 = koff; two 16B loads.
DEVINL v16h frag16(const _Float16* p) {
  v8h lo = *(const v8h*)(p);
  v8h hi = *(const v8h*)(p + 16);
  return __builtin_shufflevector(lo, hi, 0,1,2,3,4,5,6,7,8,9,10,11,12,13,14,15);
}

DEVINL v8f wmma16(v16h a, v16h b, v8f c) {
  return __builtin_amdgcn_wmma_f32_16x16x32_f16(false, a, false, b, (short)0, c,
                                                false, false);
}

// -------------------------- embedding + batchnorm --------------------------
__global__ void embed_bn_kernel(const int* __restrict__ tokens,
                                const float* __restrict__ emb,
                                const float* __restrict__ gamma,
                                const float* __restrict__ beta,
                                const float* __restrict__ mmean,
                                const float* __restrict__ mvar,
                                _Float16* __restrict__ x16) {
  int r = blockIdx.x;
  int c = threadIdx.x;
  if (c < kEMB) {
    int tok = tokens[r];
    float e = emb[(size_t)tok * kEMB + c];
    float v = (e - mmean[c]) * rsqrtf(mvar[c] + 1e-3f) * gamma[c] + beta[c];
    x16[(size_t)r * kKp1 + c] = (_Float16)v;
  }
}

// ------------------- transpose + f16 cast of weight matrices ---------------
// src: [K, N] f32 row-major -> dst: [N, LDK] f16 (K-padded, pads pre-zeroed)
__global__ void transpose_cast_kernel(const float* __restrict__ src,
                                      int K, int N, _Float16* __restrict__ dst,
                                      int ldk) {
  int idx = blockIdx.x * blockDim.x + threadIdx.x;
  if (idx >= K * N) return;
  int k = idx / N, n = idx - k * N;
  dst[(size_t)n * ldk + k] = (_Float16)src[idx];
}

// ---------------------- big parallel GEMM: C = A*B^T + bias ----------------
// A: [M, LDK] f16, Bt: [Ntiles*16, LDK] f16, C: [M, ldc] f32.
// 8 waves/block; wave w -> 16-row M tile; NT N-tiles per wave reuse the
// A fragment NT times per k-step.
template <int LDK, int NT>
__global__ void gemm_xw_kernel(const _Float16* __restrict__ A,
                               const _Float16* __restrict__ Bt,
                               const float* __restrict__ bias, int nbias,
                               float* __restrict__ C, int ldc) {
  const int lane = threadIdx.x & 31;
  const int wave = threadIdx.x >> 5;
  const int l15  = lane & 15;
  const int koff = (lane >> 4) * 8;
  const int mbase  = blockIdx.x * 128 + wave * 16;
  const int nbase0 = blockIdx.y * (16 * NT);

  const _Float16* arow = A + (size_t)(mbase + l15) * LDK + koff;
  const _Float16* brow[NT];
#pragma unroll
  for (int i = 0; i < NT; ++i)
    brow[i] = Bt + (size_t)(nbase0 + i * 16 + l15) * LDK + koff;

  v8f acc[NT];
  const v8f vzero = {};
#pragma unroll
  for (int i = 0; i < NT; ++i) acc[i] = vzero;

#pragma unroll
  for (int k = 0; k < (LDK >> 5); ++k) {
    v16h a = frag16(arow + (k << 5));
#pragma unroll
    for (int i = 0; i < NT; ++i)
      acc[i] = wmma16(a, frag16(brow[i] + (k << 5)), acc[i]);
  }

  const int rbase = mbase + 8 * (lane >> 4);
#pragma unroll
  for (int i = 0; i < NT; ++i) {
    const int n = nbase0 + i * 16 + l15;
    const float bv = (n < nbias) ? bias[n] : 0.0f;
#pragma unroll
    for (int j = 0; j < 8; ++j)
      C[(size_t)(rbase + j) * ldc + n] = acc[i][j] + bv;
  }
}

// ------------------------ persistent GRU recurrence ------------------------
// nwg workgroups, each owns 16 hidden columns. Per step: rec = h @ Ur^T via
// WMMA (Ur slice in LDS), gate math, write double-buffered h + outputs, then
// device-scope counter barrier (fence release/acquire -> global_wb/inv).
template <int LDK>
__global__ void gru_rec_kernel(const float* __restrict__ xw, int ldxw,
                               const _Float16* __restrict__ Urt,
                               const float* __restrict__ br,
                               const int* __restrict__ tokens,
                               float* h32a, float* h32b,
                               _Float16* h16a, _Float16* h16b,
                               float* __restrict__ outf,
                               _Float16* out16,
                               float* __restrict__ hfin,
                               unsigned* cnt, int u, int nwg) {
  __shared__ __align__(16) _Float16 s_ur[48 * LDK];

  const int gcol0 = blockIdx.x * 16;

  // ---- Stage this WG's 3 gate column-tiles of Ur^T into LDS (48 x LDK) ----
#if HAVE_TDM
  // Tensor Data Mover: one 2D tile DMA per gate (16 rows x LDK f16).
  // s_ur is the only __shared__ object -> LDS offset 0.
  if (threadIdx.x < 32) {
    for (int g = 0; g < 3; ++g) {
      unsigned long long ga =
          (unsigned long long)(const void*)(Urt + (size_t)(g * u + gcol0) * LDK);
      unsigned lds_off = (unsigned)(g * 16 * LDK * 2);
      // D# group 0: count=1 (user), lds_addr, global_addr[56:0], type=2.
      u32x4 g0 = {1u, lds_off, (unsigned)(ga & 0xffffffffu),
                  (unsigned)((ga >> 32) & 0x01ffffffu) | (2u << 30)};
      // D# group 1: data_size=2B; tensor_dim0=LDK; tensor_dim1=16;
      // tile_dim0=LDK; tile_dim1=16; tensor_dim0_stride=LDK;
      // tensor_dim1_stride=LDK. (bit-packed per ISA 8.4)
      i32x8 g1 = {(int)0x00010000,
                  (int)(((unsigned)LDK & 0xffffu) << 16),
                  (int)(16u << 16),
                  (int)((unsigned)LDK << 16),
                  16,
                  LDK,
                  (int)(((unsigned)LDK & 0xffffu) << 16),
                  0};
      i32x4 gz4 = {0, 0, 0, 0};
      i32x8 gz8 = {0, 0, 0, 0, 0, 0, 0, 0};
      __builtin_amdgcn_tensor_load_to_lds(g0, g1, gz4, gz4, gz8, 0);
    }
    __builtin_amdgcn_s_wait_tensorcnt(0);
  }
#else
  for (int idx = threadIdx.x; idx < 48 * LDK; idx += blockDim.x) {
    int row = idx / LDK, k = idx - row * LDK;
    int g = row >> 4, n = row & 15;
    s_ur[idx] = Urt[(size_t)(g * u + gcol0 + n) * LDK + k];
  }
#endif
  __syncthreads();

  const int lane = threadIdx.x & 31;
  const int wave = threadIdx.x >> 5;
  const int l15  = lane & 15;
  const int koff = (lane >> 4) * 8;
  const int mbase = wave * 16;                 // 8 waves x 16 = B=128 rows
  const int col   = gcol0 + l15;
  const bool colok = col < u;
  const int rbase = mbase + 8 * (lane >> 4);

  const _Float16* arowA = h16a + (size_t)(mbase + l15) * LDK + koff;
  const _Float16* arowB = h16b + (size_t)(mbase + l15) * LDK + koff;
  const _Float16* bp0 = s_ur + (size_t)(0  + l15) * LDK + koff;
  const _Float16* bp1 = s_ur + (size_t)(16 + l15) * LDK + koff;
  const _Float16* bp2 = s_ur + (size_t)(32 + l15) * LDK + koff;

  const float brz = colok ? br[col] : 0.f;
  const float brr = colok ? br[u + col] : 0.f;
  const float brh = colok ? br[2 * u + col] : 0.f;

  for (int t = 0; t < kT; ++t) {
    const bool ping = (t & 1) == 0;            // read a, write b (then swap)
    const _Float16* arow = ping ? arowA : arowB;
    const float*    h32r = ping ? h32a : h32b;
    float*          h32w = ping ? h32b : h32a;
    _Float16*       h16w = ping ? h16b : h16a;

    // Prefetch this step's xw gate rows (read-only) so the scalar gate loads
    // after the WMMA loop hit cache. Lowers to global_prefetch_b8.
    if (colok) {
#pragma unroll
      for (int j = 0; j < 8; ++j) {
        const size_t xo = ((size_t)(rbase + j) * kT + t) * (size_t)ldxw;
        __builtin_prefetch(&xw[xo + col], 0, 3);
        __builtin_prefetch(&xw[xo + u + col], 0, 3);
        __builtin_prefetch(&xw[xo + 2 * u + col], 0, 3);
      }
    }

    v8f az = {}, ar = {}, ah = {};
#pragma unroll
    for (int k = 0; k < (LDK >> 5); ++k) {
      v16h a = frag16(arow + (k << 5));
      az = wmma16(a, frag16(bp0 + (k << 5)), az);
      ar = wmma16(a, frag16(bp1 + (k << 5)), ar);
      ah = wmma16(a, frag16(bp2 + (k << 5)), ah);
    }

    if (colok) {
#pragma unroll
      for (int j = 0; j < 8; ++j) {
        const int b = rbase + j;
        const size_t rt = (size_t)b * kT + t;
        const size_t xo = rt * (size_t)ldxw;
        const float xz = xw[xo + col];
        const float xr = xw[xo + u + col];
        const float xh = xw[xo + 2 * u + col];
        const float z  = 1.f / (1.f + expf(-(xz + az[j] + brz)));
        const float r  = 1.f / (1.f + expf(-(xr + ar[j] + brr)));
        const float hh = tanhf(xh + r * (ah[j] + brh));
        const size_t ho = (size_t)b * u + col;
        const float hold = h32r[ho];
        const float hn = z * hold + (1.f - z) * hh;
        const float hs = (tokens[rt] != 0) ? hn : hold;
        h32w[ho] = hs;
        h16w[(size_t)b * LDK + col] = (_Float16)hs;
        outf[rt * (size_t)u + col] = hs;
        if (out16) out16[rt * (size_t)LDK + col] = (_Float16)hs;
      }
    }

    // ---- device-scope barrier across all nwg workgroups ----
    __builtin_amdgcn_fence(__ATOMIC_RELEASE, "agent");   // global_wb
    __syncthreads();
    if (threadIdx.x == 0) {
      __atomic_fetch_add(cnt, 1u, __ATOMIC_RELAXED);
      const unsigned target = (unsigned)(t + 1) * (unsigned)nwg;
      while (__atomic_load_n(cnt, __ATOMIC_RELAXED) < target)
        __builtin_amdgcn_s_sleep(2);
    }
    __syncthreads();
    __builtin_amdgcn_fence(__ATOMIC_ACQUIRE, "agent");   // global_inv
  }

  // Final hidden state: last write went to buffer a (T even).
  const float* hlast = (kT & 1) ? h32b : h32a;
  if (colok) {
#pragma unroll
    for (int j = 0; j < 8; ++j) {
      const int b = rbase + j;
      hfin[(size_t)b * u + col] = hlast[(size_t)b * u + col];
    }
  }
}

// ------------------------------- launcher ----------------------------------
extern "C" void kernel_launch(void* const* d_in, const int* in_sizes, int n_in,
                              void* d_out, int out_size, void* d_ws,
                              size_t ws_size, hipStream_t stream) {
  (void)in_sizes; (void)n_in; (void)out_size; (void)ws_size;

  const int*   tokens = (const int*)d_in[0];
  const float* emb    = (const float*)d_in[1];
  const float* gamma  = (const float*)d_in[2];
  const float* beta   = (const float*)d_in[3];
  const float* mmean  = (const float*)d_in[4];
  const float* mvar   = (const float*)d_in[5];
  const float* W1     = (const float*)d_in[6];
  const float* Ur1    = (const float*)d_in[7];
  const float* b1     = (const float*)d_in[8];
  const float* W2     = (const float*)d_in[9];
  const float* Ur2    = (const float*)d_in[10];
  const float* b2     = (const float*)d_in[11];

  char* ws = (char*)d_ws;
  size_t off = 0;
  auto take = [&](size_t bytes) -> void* {
    size_t o = (off + 255) & ~(size_t)255;
    off = o + bytes;
    return (void*)(ws + o);
  };

  _Float16* x16  = (_Float16*)take((size_t)kRows * kKp1 * 2);
  _Float16* w1t  = (_Float16*)take((size_t)kN1  * kKp1 * 2);
  _Float16* ur1t = (_Float16*)take((size_t)kN1  * kKp2 * 2);
  _Float16* w2t  = (_Float16*)take((size_t)kN2p * kKp2 * 2);
  _Float16* ur2t = (_Float16*)take((size_t)kN2p * kKp1 * 2);
  _Float16* o116 = (_Float16*)take((size_t)kRows * kKp2 * 2);
  float*    xw1  = (float*)take((size_t)kRows * kN1 * 4);
  float*    xw2  = (float*)take((size_t)kRows * kN2p * 4);
  // Hidden-state buffers + barrier counters packed together -> one memset.
  char* hbase = (char*)take(0);
  float*    h1a32 = (float*)take((size_t)kB * kU1 * 4);
  float*    h1b32 = (float*)take((size_t)kB * kU1 * 4);
  _Float16* h1a16 = (_Float16*)take((size_t)kB * kKp2 * 2);
  _Float16* h1b16 = (_Float16*)take((size_t)kB * kKp2 * 2);
  float*    h2a32 = (float*)take((size_t)kB * kU2 * 4);
  float*    h2b32 = (float*)take((size_t)kB * kU2 * 4);
  _Float16* h2a16 = (_Float16*)take((size_t)kB * kKp1 * 2);
  _Float16* h2b16 = (_Float16*)take((size_t)kB * kKp1 * 2);
  unsigned* cnt1  = (unsigned*)take(256);
  unsigned* cnt2  = (unsigned*)take(256);
  char* hend = ws + off;

  // Output layout: (out2, out1, h2, h1) concatenated, f32.
  float* out2 = (float*)d_out;                           // [128,256,200]
  float* out1 = out2 + (size_t)kB * kT * kU2;            // [128,256,400]
  float* h2f  = out1 + (size_t)kB * kT * kU1;            // [128,200]
  float* h1f  = h2f + (size_t)kB * kU2;                  // [128,400]

  // Zero regions whose K/N pads must be 0, plus h state + barrier counters.
  (void)hipMemsetAsync(x16,  0, (size_t)kRows * kKp1 * 2, stream);
  (void)hipMemsetAsync(w1t,  0, (size_t)kN1  * kKp1 * 2, stream);
  (void)hipMemsetAsync(ur1t, 0, (size_t)kN1  * kKp2 * 2, stream);
  (void)hipMemsetAsync(w2t,  0, (size_t)kN2p * kKp2 * 2, stream);
  (void)hipMemsetAsync(ur2t, 0, (size_t)kN2p * kKp1 * 2, stream);
  (void)hipMemsetAsync(o116, 0, (size_t)kRows * kKp2 * 2, stream);
  (void)hipMemsetAsync(hbase, 0, (size_t)(hend - hbase), stream);

  // 1) Embedding gather + BatchNorm -> f16 [32768, 224]
  embed_bn_kernel<<<kRows, 256, 0, stream>>>(tokens, emb, gamma, beta, mmean,
                                             mvar, x16);

  // 2) Transposed f16 weights
  {
    int n;
    n = kEMB * kN1;
    transpose_cast_kernel<<<(n + 255) / 256, 256, 0, stream>>>(W1, kEMB, kN1, w1t, kKp1);
    n = kU1 * kN1;
    transpose_cast_kernel<<<(n + 255) / 256, 256, 0, stream>>>(Ur1, kU1, kN1, ur1t, kKp2);
    n = kU1 * 600;
    transpose_cast_kernel<<<(n + 255) / 256, 256, 0, stream>>>(W2, kU1, 600, w2t, kKp2);
    n = kU2 * 600;
    transpose_cast_kernel<<<(n + 255) / 256, 256, 0, stream>>>(Ur2, kU2, 600, ur2t, kKp1);
  }

  // 3) xw1 = x @ W1 + b1[0] : [32768, 1200] f32   (5 N-tiles per wave)
  gemm_xw_kernel<kKp1, 5><<<dim3(kRows / 128, kN1 / 80), 256, 0, stream>>>(
      x16, w1t, b1, kN1, xw1, kN1);

  // 4) GRU1 recurrence (25 WGs x 16 cols = 400)
  gru_rec_kernel<kKp2><<<25, 256, 0, stream>>>(
      xw1, kN1, ur1t, b1 + kN1, tokens, h1a32, h1b32, h1a16, h1b16,
      out1, o116, h1f, cnt1, kU1, 25);

  // 5) xw2 = out1 @ W2 + b2[0] : [32768, 608] f32 (600 valid; 2 N-tiles/wave)
  gemm_xw_kernel<kKp2, 2><<<dim3(kRows / 128, kN2p / 32), 256, 0, stream>>>(
      o116, w2t, b2, 600, xw2, kN2p);

  // 6) GRU2 recurrence (13 WGs; last WG covers cols 192..199 only)
  gru_rec_kernel<kKp1><<<13, 256, 0, stream>>>(
      xw2, kN2p, ur2t, b2 + 600, tokens, h2a32, h2b32, h2a16, h2b16,
      out2, (_Float16*)nullptr, h2f, cnt2, kU2, 13);
}
